// IterativeFFNN_62139586838976
// MI455X (gfx1250) — matrix operand order repaired and checked by reading, not compiled, not run on hardware
//
#include <hip/hip_runtime.h>

#define BDIM 16384
#define TDIM 256
#define FDIM 19
#define PDIM 7

typedef _Float16 v16h __attribute__((ext_vector_type(16)));
typedef _Float16 v8h  __attribute__((ext_vector_type(8)));
typedef __fp16   hp2  __attribute__((ext_vector_type(2)));
typedef float    v8f  __attribute__((ext_vector_type(8)));

// ---------------- LDS layout (bytes) ----------------
// w1b [112][32]  f16 : 7168
// w2b [160][128] f16 : 40960
// w3b [112][160] f16 : 35840
// w4b [16][128]  f16 : 4096
// biases (112+160+112+16 f32) : 1600
// per-wave (x4): bufA 5632 | bufB 5632 | outS 576 | predB 256 = 12096
#define OFF_W1   0
#define OFF_W2   7168
#define OFF_W3   48128
#define OFF_W4   83968
#define OFF_BB   88064
#define OFF_WAVE 89664
#define WAVE_SZ  12096
#define SHMEM_BYTES (OFF_WAVE + 4 * WAVE_SZ)   // 138048
#define SA 176   // activation row stride in halves
#define SO 9     // outS row stride in floats (bank-conflict-free gather)

__device__ __forceinline__ void wave_lds_fence() {
    __builtin_amdgcn_wave_barrier();
    asm volatile("s_wait_dscnt 0" ::: "memory");
    __builtin_amdgcn_wave_barrier();
}

__device__ __forceinline__ float fast_tanhf(float x) {
    x = fminf(10.f, fmaxf(-10.f, x));
    float e = __expf(2.f * x);
    return __fdividef(e - 1.f, e + 1.f);
}

// C[16x16·NT] = relu(A[16xK] * B[Kx16·NT] + bias)
// Bias is pre-loaded into the WMMA accumulator (free f32 add in the matrix
// unit); epilogue is packed f16: cvt_pkrtz pairs + v_pk_max relu.
template <int NT, int KT>
__device__ __forceinline__ void layer_f16(const _Float16* aBuf, const _Float16* wB,
                                          int kpad, const float* bias, _Float16* oBuf,
                                          int mrow, int hi) {
    v16h a[KT];
    #pragma unroll
    for (int kt = 0; kt < KT; ++kt) {       // A: K[base..+7] and K[base+16..+23]
        int ab = kt * 32 + hi * 8;
        ((v8h*)&a[kt])[0] = *(const v8h*)(aBuf + mrow * SA + ab);
        ((v8h*)&a[kt])[1] = *(const v8h*)(aBuf + mrow * SA + ab + 16);
    }
    const hp2 zz = {(__fp16)0.f, (__fp16)0.f};
    #pragma unroll
    for (int nt = 0; nt < NT; ++nt) {
        float bv = bias[nt * 16 + mrow];     // bias indexed by N = lane&15
        v8f c;
        #pragma unroll
        for (int j = 0; j < 8; ++j) c[j] = bv;   // independent splats: fill hazard slots
        const _Float16* wcol = wB + (nt * 16 + mrow) * kpad;  // lane owns column N
        #pragma unroll
        for (int kt = 0; kt < KT; ++kt) {
            v16h b;
            int kb = kt * 32 + hi * 8;       // B: K[base..+7] and K[base+16..+23]
            ((v8h*)&b)[0] = *(const v8h*)(wcol + kb);
            ((v8h*)&b)[1] = *(const v8h*)(wcol + kb + 16);
            c = __builtin_amdgcn_wmma_f32_16x16x32_f16(false, a[kt], false, b,
                                                       (short)0, c, false, false);
        }
        #pragma unroll
        for (int j = 0; j < 4; ++j) {        // rows 2j,2j+1 -> one packed reg
            union { hp2 v; _Float16 s[2]; } u;
            u.v = __builtin_amdgcn_cvt_pkrtz(c[2 * j], c[2 * j + 1]);
            u.v = __builtin_elementwise_max(u.v, zz);   // v_pk_max_num_f16 relu
            oBuf[(2 * j     + hi * 8) * SA + nt * 16 + mrow] = u.s[0];
            oBuf[(2 * j + 1 + hi * 8) * SA + nt * 16 + mrow] = u.s[1];
        }
    }
}

// Per-timestep gate: cond = (sum over x[:, t, :]) != 0
__global__ void flags_kernel(const float* __restrict__ x, int* __restrict__ flags) {
    __shared__ float red[256];
    const int t = blockIdx.x;
    float s = 0.f;
    for (int b = threadIdx.x; b < BDIM; b += blockDim.x) {
        const float* p = x + (size_t)b * TDIM * FDIM + (size_t)t * FDIM;
        #pragma unroll
        for (int f = 0; f < FDIM; ++f) s += p[f];
    }
    red[threadIdx.x] = s;
    __syncthreads();
    for (int off = 128; off; off >>= 1) {
        if (threadIdx.x < off) red[threadIdx.x] += red[threadIdx.x + off];
        __syncthreads();
    }
    if (threadIdx.x == 0) flags[t] = (red[0] != 0.f) ? 1 : 0;
}

__global__ __launch_bounds__(128, 1) void ffnn_kernel(
    const float* __restrict__ x, const float* __restrict__ pred_in,
    const float* __restrict__ W1, const float* __restrict__ b1,
    const float* __restrict__ W2, const float* __restrict__ b2,
    const float* __restrict__ W3, const float* __restrict__ b3,
    const float* __restrict__ W4, const float* __restrict__ b4,
    float* __restrict__ out, const int* __restrict__ flags) {
    extern __shared__ char smem[];
    _Float16* w1b = (_Float16*)(smem + OFF_W1);
    _Float16* w2b = (_Float16*)(smem + OFF_W2);
    _Float16* w3b = (_Float16*)(smem + OFF_W3);
    _Float16* w4b = (_Float16*)(smem + OFF_W4);
    float* bb  = (float*)(smem + OFF_BB);
    float* bb1 = bb, *bb2 = bb + 112, *bb3 = bb + 272, *bb4 = bb + 384;

    const int tid  = threadIdx.x;
    const int wave = tid >> 5;
    const int lane = tid & 31;
    const int mrow = lane & 15;
    const int hi   = lane >> 4;

    char* wbase = smem + OFF_WAVE + wave * WAVE_SZ;
    _Float16* bufA  = (_Float16*)wbase;                 // 16 x SA halves
    _Float16* bufB  = (_Float16*)(wbase + 5632);
    float*    outS  = (float*)(wbase + 11264);          // 16 x SO f32
    _Float16* predB = (_Float16*)(wbase + 11840);       // 16 x 8 f16

    // ---- stage weights f32 -> f16, padded [N][Kpad], zero-filled pads ----
    for (int i = tid; i < 112 * 32;  i += 128) { int n = i >> 5, k = i & 31;
        w1b[i] = (n < 100 && k < 26)  ? (_Float16)W1[n * 26 + k]  : (_Float16)0.f; }
    for (int i = tid; i < 160 * 128; i += 128) { int n = i >> 7, k = i & 127;
        w2b[i] = (n < 150 && k < 100) ? (_Float16)W2[n * 100 + k] : (_Float16)0.f; }
    for (int i = tid; i < 112 * 160; i += 128) { int n = i / 160, k = i - n * 160;
        w3b[i] = (n < 100 && k < 150) ? (_Float16)W3[n * 150 + k] : (_Float16)0.f; }
    for (int i = tid; i < 16 * 128;  i += 128) { int n = i >> 7, k = i & 127;
        w4b[i] = (n < 7 && k < 100)   ? (_Float16)W4[n * 100 + k] : (_Float16)0.f; }
    for (int i = tid; i < 112; i += 128) bb1[i] = (i < 100) ? b1[i] : 0.f;
    for (int i = tid; i < 160; i += 128) bb2[i] = (i < 150) ? b2[i] : 0.f;
    for (int i = tid; i < 112; i += 128) bb3[i] = (i < 100) ? b3[i] : 0.f;
    for (int i = tid; i < 16;  i += 128) bb4[i] = (i < 7)   ? b4[i] : 0.f;

    // zero activation buffers (pad columns must stay 0 forever)
    for (int i = lane; i < 16 * SA; i += 32) { bufA[i] = (_Float16)0.f; bufB[i] = (_Float16)0.f; }

    const int bRow = blockIdx.x * 64 + wave * 16 + mrow;  // row owned by lanes 0..15
    if (lane < 16) {
        #pragma unroll
        for (int c = 0; c < 8; ++c)
            predB[lane * 8 + c] = (c < 7) ? (_Float16)pred_in[(size_t)bRow * 7 + c]
                                          : (_Float16)0.f;
    }
    __syncthreads();

    float outReg[7];
    #pragma unroll
    for (int c = 0; c < 7; ++c) outReg[c] = 0.f;

    const size_t xRow = (size_t)TDIM * FDIM;

    for (int t = 0; t < TDIM; ++t) {
        if (flags[t] == 0) continue;  // uniform: out & pred unchanged -> skip exactly

        // ---- stage A0 = [x(19) | pred(7) | 0(6)] as 16x32 f16; lane owns column k ----
        const int k = lane;
        #pragma unroll 4
        for (int m = 0; m < 16; ++m) {
            _Float16 hv;
            if (k < FDIM) {
                const float* xp = x + (size_t)(blockIdx.x * 64 + wave * 16 + m) * xRow
                                    + (size_t)t * FDIM;
                hv = (_Float16)xp[k];
                if (k == 0 && t + 1 < TDIM) __builtin_prefetch(xp + FDIM, 0, 1);
            } else if (k < 26) {
                hv = predB[m * 8 + (k - 19)];
            } else {
                hv = (_Float16)0.f;
            }
            bufA[m * SA + k] = hv;
        }
        wave_lds_fence();

        layer_f16<7, 1>(bufA, w1b, 32,  bb1, bufB, mrow, hi);   // 26->100 (112)
        wave_lds_fence();
        layer_f16<10, 4>(bufB, w2b, 128, bb2, bufA, mrow, hi);  // 100->150 (160)
        wave_lds_fence();
        layer_f16<7, 5>(bufA, w3b, 160, bb3, bufB, mrow, hi);   // 150->100 (112)
        wave_lds_fence();

        // ---- layer 4: [16x128] x [128x16] -> tanh (bias pre-loaded in C) ----
        {
            v16h a[4];
            #pragma unroll
            for (int kt = 0; kt < 4; ++kt) {
                int ab = kt * 32 + hi * 8;
                ((v8h*)&a[kt])[0] = *(const v8h*)(bufB + mrow * SA + ab);
                ((v8h*)&a[kt])[1] = *(const v8h*)(bufB + mrow * SA + ab + 16);
            }
            float bv = bb4[mrow];
            v8f c;
            #pragma unroll
            for (int j = 0; j < 8; ++j) c[j] = bv;
            const _Float16* wcol = w4b + mrow * 128;
            #pragma unroll
            for (int kt = 0; kt < 4; ++kt) {
                v16h b;
                int kb = kt * 32 + hi * 8;
                ((v8h*)&b)[0] = *(const v8h*)(wcol + kb);
                ((v8h*)&b)[1] = *(const v8h*)(wcol + kb + 16);
                c = __builtin_amdgcn_wmma_f32_16x16x32_f16(false, a[kt], false, b,
                                                           (short)0, c, false, false);
            }
            if (mrow < 7) {
                #pragma unroll
                for (int j = 0; j < 8; ++j)
                    outS[(j + hi * 8) * SO + mrow] = fast_tanhf(c[j]);
            }
        }
        wave_lds_fence();

        // ---- per-row softmax feedback; lanes 0..15 each own one batch row ----
        if (lane < 16) {
            float v[7], vmax = -1e30f, s = 0.f;
            #pragma unroll
            for (int c2 = 0; c2 < 7; ++c2) {
                v[c2] = outS[lane * SO + c2];
                outReg[c2] = v[c2];
                vmax = fmaxf(vmax, v[c2]);
            }
            #pragma unroll
            for (int c2 = 0; c2 < 7; ++c2) { v[c2] = __expf(v[c2] - vmax); s += v[c2]; }
            float inv = __fdividef(1.f, s);
            #pragma unroll
            for (int c2 = 0; c2 < 7; ++c2) predB[lane * 8 + c2] = (_Float16)(v[c2] * inv);
        }
        wave_lds_fence();
    }

    if (lane < 16) {
        #pragma unroll
        for (int c2 = 0; c2 < 7; ++c2)
            out[(size_t)bRow * 7 + c2] = outReg[c2];
    }
}

extern "C" void kernel_launch(void* const* d_in, const int* in_sizes, int n_in,
                              void* d_out, int out_size, void* d_ws, size_t ws_size,
                              hipStream_t stream) {
    const float* x       = (const float*)d_in[0];
    const float* pred_in = (const float*)d_in[1];
    const float* W1 = (const float*)d_in[2];
    const float* b1 = (const float*)d_in[3];
    const float* W2 = (const float*)d_in[4];
    const float* b2 = (const float*)d_in[5];
    const float* W3 = (const float*)d_in[6];
    const float* b3 = (const float*)d_in[7];
    const float* W4 = (const float*)d_in[8];
    const float* b4 = (const float*)d_in[9];
    float* out = (float*)d_out;
    int* flags = (int*)d_ws;

    flags_kernel<<<TDIM, 256, 0, stream>>>(x, flags);

    (void)hipFuncSetAttribute((const void*)ffnn_kernel,
                              hipFuncAttributeMaxDynamicSharedMemorySize, SHMEM_BYTES);
    ffnn_kernel<<<BDIM / 64, 128, SHMEM_BYTES, stream>>>(
        x, pred_in, W1, b1, W2, b2, W3, b3, W4, b4, out, flags);
}
